// BlockWiseAttention_3968549782362
// MI455X (gfx1250) — compile-verified, matrix operand order checked by reading, not run on hardware
//
#include <hip/hip_runtime.h>
#include <hip/hip_bf16.h>
#include <math.h>
#include <stdint.h>

typedef __attribute__((ext_vector_type(16))) __bf16 v16bf;
typedef __attribute__((ext_vector_type(8)))  __bf16 v8bf;
typedef __attribute__((ext_vector_type(8)))  float  v8f;
typedef __attribute__((ext_vector_type(4)))  float  v4f;
typedef __attribute__((ext_vector_type(4)))  int    v4i;

#define NHEADS 16
#define DHEAD  64
#define CDIM   1024
#define NTOK   2048   // S + 1 global token
#define SEQ    2047
#define NBATCH 2
#define LDSK   72     // padded LDS row stride (bf16 elems) for a K=64 slab

// ---------------------------------------------------------------------------
// WMMA helpers.
// A-operand (16x32 bf16 rows): lane L holds row m=L%16, element e holds
// K = (e&7) + 8*(L/16) + 16*(e>>3)  ->  two b128 loads at +8*half / +16+8*half.
// B-operand (32x16, column-major) uses the identical mapping with lane = col.
// ---------------------------------------------------------------------------
__device__ __forceinline__ v16bf load_op(const __bf16* p, int half) {
  v8bf lo = *(const v8bf*)(p + 8 * half);
  v8bf hi = *(const v8bf*)(p + 16 + 8 * half);
  v16bf r;
#pragma unroll
  for (int i = 0; i < 8; ++i) { r[i] = lo[i]; r[8 + i] = hi[i]; }
  return r;
}

__device__ __forceinline__ v8f wmma_bf(v16bf a, v16bf b, v8f c) {
  return __builtin_amdgcn_wmma_f32_16x16x32_bf16(
      false, a, false, b, (short)0, c, false, false);
}

// ---------------------------------------------------------------------------
// Global -> LDS staging. Prefer gfx1250 async direct-to-LDS (ASYNCcnt path);
// fall back to VGPR round-trip copy if the builtin is unavailable.
// Stage a 128x64 bf16 tile (row-major, row stride CDIM) into LDS (stride LDSK).
// 256 threads: thread t handles half a row (32 elems = 64 B = 4 x b128).
// ---------------------------------------------------------------------------
#if defined(__has_builtin)
#if __has_builtin(__builtin_amdgcn_global_load_async_to_lds_b128)
#define USE_ASYNC_LDS 1
#endif
#endif

#ifdef USE_ASYNC_LDS
typedef __attribute__((address_space(1))) v4i g_v4i;   // global int4
typedef __attribute__((address_space(3))) v4i l_v4i;   // LDS int4
#endif

__device__ __forceinline__ void wait_lds_ready() {
#ifdef USE_ASYNC_LDS
#if __has_builtin(__builtin_amdgcn_s_wait_asynccnt)
  __builtin_amdgcn_s_wait_asynccnt(0);
#else
  asm volatile("s_wait_asynccnt 0x0" ::: "memory");
#endif
#endif
}

__device__ __forceinline__ void stage_tile(const __bf16* __restrict__ g,
                                           int grow_base, int gcol,
                                           __bf16* lds, int t) {
  const int row  = t >> 1;
  const int part = (t & 1) * 32;
  const __bf16* src = g + (size_t)(grow_base + row) * CDIM + gcol + part;
  __bf16* dst = lds + row * LDSK + part;
#ifdef USE_ASYNC_LDS
  // Same 16B stride on both sides -> one base pair + imm offsets (ISA adds
  // INST_OFFSET to both the global and the LDS address). Generic LDS addresses
  // truncate to addr[31:0] (ISA aperture rule), giving the AS3 offset.
  g_v4i* gp = (g_v4i*)(uintptr_t)src;
  l_v4i* lp = (l_v4i*)(uint32_t)(uintptr_t)dst;
#pragma unroll
  for (int u = 0; u < 4; ++u)
    __builtin_amdgcn_global_load_async_to_lds_b128(gp + u, lp + u, 0, 0);
#else
#pragma unroll
  for (int u = 0; u < 4; ++u)
    ((v8bf*)dst)[u] = ((const v8bf*)src)[u];
#endif
}

// ---------------------------------------------------------------------------
// fp32 -> bf16 prep (x8 vectorized); prepend the global token.
// ---------------------------------------------------------------------------
__global__ void k_prep_xg(const float* __restrict__ x,
                          const float* __restrict__ g,
                          __bf16* __restrict__ xg, int total8) {
  int i = blockIdx.x * blockDim.x + threadIdx.x;
  if (i >= total8) return;
  int e = i * 8;
  int c = e & (CDIM - 1);
  int n = (e >> 10) & (NTOK - 1);
  int b = e >> 21;                               // NTOK*CDIM == 2^21
  const float* src = (n == 0) ? (g + c)
                              : (x + ((size_t)b * SEQ + (n - 1)) * CDIM + c);
  v8bf pk;
#pragma unroll
  for (int u = 0; u < 8; ++u) pk[u] = (__bf16)src[u];
  *(v8bf*)(xg + e) = pk;
}

__global__ void k_cvt(const float* __restrict__ src,
                      __bf16* __restrict__ dst, int n8) {
  int i = blockIdx.x * blockDim.x + threadIdx.x;
  if (i >= n8) return;
  int e = i * 8;
  v8bf pk;
#pragma unroll
  for (int u = 0; u < 8; ++u) pk[u] = (__bf16)src[e + u];
  *(v8bf*)(dst + e) = pk;
}

// ---------------------------------------------------------------------------
// QKV GEMM: xg(4096x1024) @ qkv_w^T -> q/k [b,h,n,d] (q pre-scaled 1/8), vT [b,h,d,n].
// 128x128 block tile via LDS (K-step 64), 8 waves each 32(rows)x64(cols).
// TR is a COMPILE-TIME parameter: TR=true (q/k features) computes C^T by
// swapping the WMMA operands so the epilogue stores contiguous [n,d] packs;
// TR=false (v features) computes C and stores contiguous vT [d,n] packs.
// ---------------------------------------------------------------------------
template <bool TR>
__global__ void __launch_bounds__(256)
k_qkv_t(const __bf16* __restrict__ xg, const __bf16* __restrict__ wq,
        __bf16* __restrict__ qb, __bf16* __restrict__ kb,
        __bf16* __restrict__ vT, int col_off) {
  __shared__ __bf16 lA[128 * LDSK];
  __shared__ __bf16 lB[128 * LDSK];
  const int t      = threadIdx.x;
  const int lane   = t & 31;
  const int wave   = t >> 5;
  const int lane16 = lane & 15;
  const int half   = lane >> 4;
  const int wr     = wave >> 1;                  // 0..3  (32-row strip)
  const int wc     = wave & 1;                   // 0..1  (64-col strip)
  const int row_blk = blockIdx.x * 128;
  const int col_blk = col_off + blockIdx.y * 128;

  v8f acc[2][4] = {};
  for (int kk = 0; kk < CDIM; kk += 64) {
    __syncthreads();
    stage_tile(xg, row_blk, kk, lA, t);
    stage_tile(wq, col_blk, kk, lB, t);
    wait_lds_ready();
    __syncthreads();
#pragma unroll
    for (int c = 0; c < 2; ++c) {
      v16bf a0 = load_op(lA + (wr * 32 +      lane16) * LDSK + c * 32, half);
      v16bf a1 = load_op(lA + (wr * 32 + 16 + lane16) * LDSK + c * 32, half);
#pragma unroll
      for (int j = 0; j < 4; ++j) {
        v16bf bo = load_op(lB + (wc * 64 + j * 16 + lane16) * LDSK + c * 32, half);
        if constexpr (TR) {
          acc[0][j] = wmma_bf(bo, a0, acc[0][j]);   // C^T: rows=features
          acc[1][j] = wmma_bf(bo, a1, acc[1][j]);
        } else {
          acc[0][j] = wmma_bf(a0, bo, acc[0][j]);   // C: rows=tokens
          acc[1][j] = wmma_bf(a1, bo, acc[1][j]);
        }
      }
    }
  }

#pragma unroll
  for (int i = 0; i < 2; ++i)
#pragma unroll
    for (int j = 0; j < 4; ++j) {
      const int fb = col_blk + wc * 64 + j * 16;   // tile feature base (16-aligned)
      const int hh = (fb >> 6) & 15;
      if constexpr (TR) {
        // lane = token col, rows r = features fb + r + 8*half (one head)
        const int which = fb >> 10;                // 0=q, 1=k
        const int tok = row_blk + wr * 32 + i * 16 + lane16;
        const int b   = tok >> 11;
        const int n   = tok & (NTOK - 1);
        const int d0  = (fb & 63) + 8 * half;
        __bf16* dst = (which == 0) ? qb : kb;
        const float sc = (which == 0) ? 0.125f : 1.0f;  // fold 1/sqrt(64) into q
        v8bf pk;
#pragma unroll
        for (int r = 0; r < 8; ++r) pk[r] = (__bf16)(acc[i][j][r] * sc);
        *(v8bf*)(dst + ((size_t)(b * NHEADS + hh) * NTOK + n) * DHEAD + d0) = pk;
      } else {
        // v region: lane = feature col, rows r = tokens (contiguous) -> vT pack
        const int f  = fb + lane16;
        const int d  = f & 63;
        const int nb = row_blk + wr * 32 + i * 16 + 8 * half;
        const int b  = nb >> 11;
        const int n0 = nb & (NTOK - 1);
        v8bf pk;
#pragma unroll
        for (int r = 0; r < 8; ++r) pk[r] = (__bf16)acc[i][j][r];
        *(v8bf*)(vT + ((size_t)(b * NHEADS + hh) * DHEAD + d) * NTOK + n0) = pk;
      }
    }
}

// ---------------------------------------------------------------------------
// Flash attention, block-causal (64-token blocks). One wave per 32 queries
// (2 query tiles) keeps the kernel under 256 VGPRs (no s_set_vgpr_msb churn);
// the two waves of each 64-query block sit in the same workgroup so their
// shared K/V stream hits the WGP cache. Computes S^T = K*Q^T so softmax
// reduces within-lane; exp'd P^T repacks directly into the B-operand of
// O^T = V^T * P^T with zero cross-lane traffic.
// ---------------------------------------------------------------------------
__global__ void __launch_bounds__(128)
k_attn(const __bf16* __restrict__ qb, const __bf16* __restrict__ kb,
       const __bf16* __restrict__ vT, __bf16* __restrict__ attn) {
  const int lane   = threadIdx.x & 31;
  const int wave   = threadIdx.x >> 5;
  const int lane16 = lane & 15;
  const int half   = lane >> 4;
  const int wg    = blockIdx.x * 4 + wave;       // 0..2047
  const int qhalf = wg & 1;                      // 32-query half of the block
  const int qblk  = (wg >> 1) & 31;              // 64-query causal block
  const int bh    = wg >> 6;                     // 0..31
  const int b     = bh >> 4;
  const int h     = bh & 15;
  const int q0    = qblk * 64 + qhalf * 32;
  const int kmax  = (qblk + 1) * 64;             // exact block-causal bound

  v16bf qop[2][2];
#pragma unroll
  for (int qt = 0; qt < 2; ++qt) {
    const __bf16* qrow = qb + ((size_t)bh * NTOK + q0 + qt * 16 + lane16) * DHEAD;
    qop[qt][0] = load_op(qrow, half);
    qop[qt][1] = load_op(qrow + 32, half);
  }

  v8f o[2][4] = {};                              // [qtile][dtile] O^T accumulators
  float mrun[2], lrun[2];
#pragma unroll
  for (int qt = 0; qt < 2; ++qt) { mrun[qt] = -INFINITY; lrun[qt] = 0.f; }

  for (int kc = 0; kc < kmax; kc += 32) {
    const __bf16* kr0 = kb + ((size_t)bh * NTOK + kc + lane16) * DHEAD;
    const __bf16* kr1 = kr0 + 16 * DHEAD;
    v16bf ka00 = load_op(kr0, half),  ka01 = load_op(kr0 + 32, half);
    v16bf ka10 = load_op(kr1, half),  ka11 = load_op(kr1 + 32, half);
    v16bf va[4];
#pragma unroll
    for (int dt = 0; dt < 4; ++dt)
      va[dt] = load_op(vT + ((size_t)bh * DHEAD + dt * 16 + lane16) * NTOK + kc, half);

    // Both query tiles' score WMMAs first: 8 independent matrix ops give the
    // scheduler room to cover the WMMA->VALU hazard before the max-reduce.
    v8f s[2][2];
#pragma unroll
    for (int qt = 0; qt < 2; ++qt) {
      s[qt][0] = wmma_bf(ka00, qop[qt][0], v8f{});
      s[qt][1] = wmma_bf(ka10, qop[qt][0], v8f{});
      s[qt][0] = wmma_bf(ka01, qop[qt][1], s[qt][0]);
      s[qt][1] = wmma_bf(ka11, qop[qt][1], s[qt][1]);
    }

#pragma unroll
    for (int qt = 0; qt < 2; ++qt) {
      float cm = -INFINITY;
#pragma unroll
      for (int r = 0; r < 8; ++r) { cm = fmaxf(cm, s[qt][0][r]); cm = fmaxf(cm, s[qt][1][r]); }
      cm = fmaxf(cm, __shfl_xor(cm, 16, 32));    // combine key-halves
      float mnew  = fmaxf(mrun[qt], cm);
      float alpha = __expf(mrun[qt] - mnew);

      float ls = 0.f;
      v16bf pb;                                  // B-operand: keys kc..kc+31
#pragma unroll
      for (int r = 0; r < 8; ++r) {
        float p0 = __expf(s[qt][0][r] - mnew);
        float p1 = __expf(s[qt][1][r] - mnew);
        ls += p0 + p1;
        pb[r]     = (__bf16)p0;
        pb[r + 8] = (__bf16)p1;
      }
      mrun[qt] = mnew;
      lrun[qt] = lrun[qt] * alpha + ls;

#pragma unroll
      for (int dt = 0; dt < 4; ++dt) {
#pragma unroll
        for (int r = 0; r < 8; ++r) o[qt][dt][r] *= alpha;
        o[qt][dt] = wmma_bf(va[dt], pb, o[qt][dt]);
      }
    }
  }

#pragma unroll
  for (int qt = 0; qt < 2; ++qt) {
    float lt  = lrun[qt] + __shfl_xor(lrun[qt], 16, 32);
    float inv = 1.0f / lt;
#pragma unroll
    for (int dt = 0; dt < 4; ++dt) {
      v8bf pk;
#pragma unroll
      for (int r = 0; r < 8; ++r) pk[r] = (__bf16)(o[qt][dt][r] * inv);
      *(v8bf*)(attn + ((size_t)b * NTOK + q0 + qt * 16 + lane16) * CDIM +
               h * DHEAD + dt * 16 + 8 * half) = pk;
    }
  }
}

// ---------------------------------------------------------------------------
// Proj GEMM: attn @ proj_w^T + bias, drop token 0, fp32 out.
// Same LDS-tiled structure; always computes C^T so stores are 2x float4.
// ---------------------------------------------------------------------------
__global__ void __launch_bounds__(256)
k_proj(const __bf16* __restrict__ attn, const __bf16* __restrict__ wp,
       const float* __restrict__ bias, float* __restrict__ out) {
  __shared__ __bf16 lA[128 * LDSK];
  __shared__ __bf16 lB[128 * LDSK];
  const int t      = threadIdx.x;
  const int lane   = t & 31;
  const int wave   = t >> 5;
  const int lane16 = lane & 15;
  const int half   = lane >> 4;
  const int wr     = wave >> 1;
  const int wc     = wave & 1;
  const int row_blk = blockIdx.x * 128;
  const int col_blk = blockIdx.y * 128;

  v8f acc[2][4] = {};
  for (int kk = 0; kk < CDIM; kk += 64) {
    __syncthreads();
    stage_tile(attn, row_blk, kk, lA, t);
    stage_tile(wp, col_blk, kk, lB, t);
    wait_lds_ready();
    __syncthreads();
#pragma unroll
    for (int c = 0; c < 2; ++c) {
      v16bf a0 = load_op(lA + (wr * 32 +      lane16) * LDSK + c * 32, half);
      v16bf a1 = load_op(lA + (wr * 32 + 16 + lane16) * LDSK + c * 32, half);
#pragma unroll
      for (int j = 0; j < 4; ++j) {
        v16bf bo = load_op(lB + (wc * 64 + j * 16 + lane16) * LDSK + c * 32, half);
        acc[0][j] = wmma_bf(bo, a0, acc[0][j]);  // C^T: rows=features, cols=tokens
        acc[1][j] = wmma_bf(bo, a1, acc[1][j]);
      }
    }
  }

#pragma unroll
  for (int i = 0; i < 2; ++i) {
    const int tok = row_blk + wr * 32 + i * 16 + lane16;
    const int b   = tok >> 11;
    const int n   = tok & (NTOK - 1);
    if (n == 0) continue;                        // dropped global token row
    float* orow = out + ((size_t)b * SEQ + (n - 1)) * CDIM;
#pragma unroll
    for (int j = 0; j < 4; ++j) {
      const int f0 = col_blk + wc * 64 + j * 16 + 8 * half;
      const v4f* bv = (const v4f*)(bias + f0);
      v4f lo, hi;
#pragma unroll
      for (int r = 0; r < 4; ++r) { lo[r] = acc[i][j][r] + bv[0][r];
                                    hi[r] = acc[i][j][r + 4] + bv[1][r]; }
      *(v4f*)(orow + f0)     = lo;
      *(v4f*)(orow + f0 + 4) = hi;
    }
  }
}

// ---------------------------------------------------------------------------
extern "C" void kernel_launch(void* const* d_in, const int* in_sizes, int n_in,
                              void* d_out, int out_size, void* d_ws, size_t ws_size,
                              hipStream_t stream) {
  (void)in_sizes; (void)n_in; (void)out_size; (void)ws_size;
  const float* x    = (const float*)d_in[0];
  const float* gtok = (const float*)d_in[1];
  const float* qkvw = (const float*)d_in[2];
  const float* projw= (const float*)d_in[3];
  const float* projb= (const float*)d_in[4];
  float* out = (float*)d_out;

  const size_t MB = 1u << 20;
  char* w = (char*)d_ws;
  __bf16* xg   = (__bf16*)(w);             // 8 MB
  __bf16* wqb  = (__bf16*)(w + 8  * MB);   // 6 MB
  __bf16* wpb  = (__bf16*)(w + 14 * MB);   // 2 MB
  __bf16* qb   = (__bf16*)(w + 16 * MB);   // 8 MB [b,h,n,d]
  __bf16* kb   = (__bf16*)(w + 24 * MB);   // 8 MB [b,h,n,d]
  __bf16* vT   = (__bf16*)(w + 32 * MB);   // 8 MB [b,h,d,n]
  __bf16* attn = (__bf16*)(w + 40 * MB);   // 8 MB [b,n,c]

  k_prep_xg<<<2048, 256, 0, stream>>>(x, gtok, xg, NBATCH * NTOK * CDIM / 8);
  k_cvt<<<1536, 256, 0, stream>>>(qkvw, wqb, 3 * CDIM * CDIM / 8);
  k_cvt<<<512, 256, 0, stream>>>(projw, wpb, CDIM * CDIM / 8);

  // QKV GEMM split into compile-time orientations (branch-free inner loops):
  k_qkv_t<true ><<<dim3(32, 16), 256, 0, stream>>>(xg, wqb, qb, kb, vT, 0);    // q/k
  k_qkv_t<false><<<dim3(32, 8 ), 256, 0, stream>>>(xg, wqb, qb, kb, vT, 2048); // v

  k_attn<<<512, 128, 0, stream>>>(qb, kb, vT, attn);              // 2048 waves
  k_proj<<<dim3(32, 8), 256, 0, stream>>>(attn, wpb, projb, out); // 4096x1024x1024
}
